// LSTMModel_46291157517097
// MI455X (gfx1250) — matrix operand order repaired and checked by reading, not compiled, not run on hardware
//
#include <hip/hip_runtime.h>
#include <hip/hip_bf16.h>
#include <math.h>

// ---------------- problem constants ----------------
#define B_   64
#define T_   512
#define I_   128
#define H_   512
#define G_   2048          // 4*H
#define OUT_ 24

#define NWG  32            // persistent grid size (must all be co-resident)
#define BDIM 256           // 8 wave32s
#define LDP  68            // LDS tile row pitch (floats), padded vs 64 banks

typedef __bf16 bf16_t;
typedef __attribute__((ext_vector_type(16))) __bf16 v16bf;
typedef __attribute__((ext_vector_type(8)))  __bf16 v8bf;
typedef __attribute__((ext_vector_type(8)))  float  v8f;

// ---------------- WMMA fragment helpers (layouts per CDNA5 ISA 7.12.2) ----

// A-matrix 16x32 bf16, lane L: row M=L&15, elems 0..7 -> K=half*8+0..7,
// elems 8..15 -> K=16+half*8+0..7  (half = L>>4). Two contiguous 16B loads.
__device__ inline v16bf load_frag_a(const bf16_t* __restrict__ A, int lda,
                                    int m0, int k0, int lane) {
    const int r    = lane & 15;
    const int half = lane >> 4;
    const bf16_t* p = A + (size_t)(m0 + r) * lda + k0 + half * 8;
    v8bf lo = *(const v8bf*)(p);        // K = k0+half*8    .. +7
    v8bf hi = *(const v8bf*)(p + 16);   // K = k0+16+half*8 .. +7
    v16bf o;
#pragma unroll
    for (int i = 0; i < 8; ++i) { o[i] = lo[i]; o[i + 8] = hi[i]; }
    return o;
}

// B-matrix 32x16 bf16 (K x N), B[k][n] = W[n][k], W row-major (N x K):
// lane n holds K = k0+half*8+0..7 (elems 0..7), K = k0+16+half*8+0..7 (elems 8..15).
__device__ inline v16bf load_frag_b(const bf16_t* __restrict__ W, int ldw,
                                    int n0, int k0, int lane) {
    const int n    = lane & 15;
    const int half = lane >> 4;
    const bf16_t* p = W + (size_t)(n0 + n) * ldw + k0 + half * 8;
    v8bf lo = *(const v8bf*)(p);
    v8bf hi = *(const v8bf*)(p + 16);
    v16bf o;
#pragma unroll
    for (int i = 0; i < 8; ++i) { o[i] = lo[i]; o[i + 8] = hi[i]; }
    return o;
}

__device__ inline v8f wmma_bf16(v16bf a, v16bf b, v8f c) {
    return __builtin_amdgcn_wmma_f32_16x16x32_bf16(
        /*neg_a=*/false, a, /*neg_b=*/false, b,
        /*c_mod=*/(short)0, c, /*reuse_a=*/false, /*reuse_b=*/false);
}

// C/D 16x16 f32 -> LDS tile: VGPR r, lanes 0-15: M=r, N=lane;
// lanes 16-31: M=r+8, N=lane-16.
__device__ inline void store_acc_lds(float* __restrict__ tile,
                                     int m0, int n0, v8f acc, int lane) {
    const int n    = lane & 15;
    const int half = lane >> 4;
    float* p = tile + (size_t)(m0 + half * 8) * LDP + n0 + n;
#pragma unroll
    for (int r = 0; r < 8; ++r) p[(size_t)r * LDP] = acc[r];
}

// ---------------- grid barrier (agent-scope atomics) ----------------------
__device__ inline void grid_barrier(unsigned* bar) {
    __syncthreads();
    if (threadIdx.x == 0) {
        __threadfence();   // release my global writes
        unsigned gen = __hip_atomic_load(&bar[1], __ATOMIC_ACQUIRE,
                                         __HIP_MEMORY_SCOPE_AGENT);
        unsigned prev = __hip_atomic_fetch_add(&bar[0], 1u, __ATOMIC_ACQ_REL,
                                               __HIP_MEMORY_SCOPE_AGENT);
        if (prev == NWG - 1u) {
            __hip_atomic_store(&bar[0], 0u, __ATOMIC_RELAXED,
                               __HIP_MEMORY_SCOPE_AGENT);
            __hip_atomic_fetch_add(&bar[1], 1u, __ATOMIC_RELEASE,
                                   __HIP_MEMORY_SCOPE_AGENT);
        } else {
            while (__hip_atomic_load(&bar[1], __ATOMIC_ACQUIRE,
                                     __HIP_MEMORY_SCOPE_AGENT) == gen) {
                __builtin_amdgcn_s_sleep(2);
            }
        }
    }
    __syncthreads();
    __threadfence();       // acquire other WGs' writes
}

// ---------------- fused 2-layer persistent LSTM ---------------------------
// Weights are pre-permuted: new row n' = w*64 + gate*16 + q  maps old row
// gate*512 + w*16 + q, so WG w's 64 output columns hold all 4 gates of its
// 16 h-columns -> gate nonlinearity needs only a workgroup barrier + LDS.
__global__ void __launch_bounds__(BDIM)
lstm_persistent(const bf16_t* __restrict__ x,      // (B,T,I) bf16
                const bf16_t* __restrict__ Wih0p,  // (2048,128) permuted
                const bf16_t* __restrict__ Whh0p,  // (2048,512) permuted
                const bf16_t* __restrict__ Wih1p,  // (2048,512) permuted
                const bf16_t* __restrict__ Whh1p,  // (2048,512) permuted
                const float*  __restrict__ bias0p, // (2048) permuted bih+bhh
                const float*  __restrict__ bias1p,
                bf16_t* __restrict__ h0, float* __restrict__ c0,
                bf16_t* __restrict__ h1, float* __restrict__ c1,
                unsigned* __restrict__ bar) {
    __shared__ float tile[64 * LDP];               // 64x64 gate tile (+pad)

    const int wg   = blockIdx.x;
    const int tid  = threadIdx.x;
    const int wave = tid >> 5;
    const int lane = tid & 31;
    const int m0   = (wave & 3) * 16;              // M tile (4 per WG)
    const int nl   = (wave >> 2) * 32;             // local N base (0 or 32)
    const int nG   = wg * 64 + nl;                 // permuted global N base

    for (int t = 0; t < T_; ++t) {
        // ======== layer 0: g = x_t @ Wih0p^T + h0 @ Whh0p^T ========
        {
            v8f acc0 = {}; v8f acc1 = {};
            const bf16_t* xt = x + (size_t)t * I_;  // row b at stride T*I
#pragma unroll 2
            for (int kk = 0; kk < I_; kk += 32) {
                v16bf a  = load_frag_a(xt, T_ * I_, m0, kk, lane);
                v16bf b0 = load_frag_b(Wih0p, I_, nG,      kk, lane);
                v16bf b1 = load_frag_b(Wih0p, I_, nG + 16, kk, lane);
                acc0 = wmma_bf16(a, b0, acc0);
                acc1 = wmma_bf16(a, b1, acc1);
            }
#pragma unroll 2
            for (int kk = 0; kk < H_; kk += 32) {
                v16bf a  = load_frag_a(h0, H_, m0, kk, lane);
                v16bf b0 = load_frag_b(Whh0p, H_, nG,      kk, lane);
                v16bf b1 = load_frag_b(Whh0p, H_, nG + 16, kk, lane);
                acc0 = wmma_bf16(a, b0, acc0);
                acc1 = wmma_bf16(a, b1, acc1);
            }
            store_acc_lds(tile, m0, nl,      acc0, lane);
            store_acc_lds(tile, m0, nl + 16, acc1, lane);
        }
        __syncthreads();
        // gates for this WG's 16 h-columns (all 4 gates are in our tile)
#pragma unroll
        for (int it = 0; it < 4; ++it) {
            const int idx = it * BDIM + tid;       // 0..1023
            const int b   = idx >> 4;              // batch row
            const int q   = idx & 15;              // h-col within slice
            const float* tr = tile + (size_t)b * LDP;
            float gi = tr[q]      + bias0p[wg * 64 + q];
            float gf = tr[16 + q] + bias0p[wg * 64 + 16 + q];
            float gg = tr[32 + q] + bias0p[wg * 64 + 32 + q];
            float go = tr[48 + q] + bias0p[wg * 64 + 48 + q];
            float iv = 1.f / (1.f + __expf(-gi));
            float fv = 1.f / (1.f + __expf(-gf));
            float gv = tanhf(gg);
            float ov = 1.f / (1.f + __expf(-go));
            const int j = wg * 16 + q;             // true h column
            float cv = fv * c0[(size_t)b * H_ + j] + iv * gv;
            c0[(size_t)b * H_ + j] = cv;
            h0[(size_t)b * H_ + j] = (bf16_t)(ov * tanhf(cv));
        }
        grid_barrier(bar);                          // publish h0 to all WGs

        // ======== layer 1: g = h0 @ Wih1p^T + h1 @ Whh1p^T ========
        {
            v8f acc0 = {}; v8f acc1 = {};
#pragma unroll 2
            for (int kk = 0; kk < H_; kk += 32) {
                v16bf a  = load_frag_a(h0, H_, m0, kk, lane);
                v16bf b0 = load_frag_b(Wih1p, H_, nG,      kk, lane);
                v16bf b1 = load_frag_b(Wih1p, H_, nG + 16, kk, lane);
                acc0 = wmma_bf16(a, b0, acc0);
                acc1 = wmma_bf16(a, b1, acc1);
            }
#pragma unroll 2
            for (int kk = 0; kk < H_; kk += 32) {
                v16bf a  = load_frag_a(h1, H_, m0, kk, lane);
                v16bf b0 = load_frag_b(Whh1p, H_, nG,      kk, lane);
                v16bf b1 = load_frag_b(Whh1p, H_, nG + 16, kk, lane);
                acc0 = wmma_bf16(a, b0, acc0);
                acc1 = wmma_bf16(a, b1, acc1);
            }
            store_acc_lds(tile, m0, nl,      acc0, lane);
            store_acc_lds(tile, m0, nl + 16, acc1, lane);
        }
        __syncthreads();
#pragma unroll
        for (int it = 0; it < 4; ++it) {
            const int idx = it * BDIM + tid;
            const int b   = idx >> 4;
            const int q   = idx & 15;
            const float* tr = tile + (size_t)b * LDP;
            float gi = tr[q]      + bias1p[wg * 64 + q];
            float gf = tr[16 + q] + bias1p[wg * 64 + 16 + q];
            float gg = tr[32 + q] + bias1p[wg * 64 + 32 + q];
            float go = tr[48 + q] + bias1p[wg * 64 + 48 + q];
            float iv = 1.f / (1.f + __expf(-gi));
            float fv = 1.f / (1.f + __expf(-gf));
            float gv = tanhf(gg);
            float ov = 1.f / (1.f + __expf(-go));
            const int j = wg * 16 + q;
            float cv = fv * c1[(size_t)b * H_ + j] + iv * gv;
            c1[(size_t)b * H_ + j] = cv;
            h1[(size_t)b * H_ + j] = (bf16_t)(ov * tanhf(cv));
        }
        grid_barrier(bar);                          // publish h1 to all WGs
    }
}

// ---------------- prep + epilogue kernels ---------------------------------
__global__ void f32_to_bf16_kernel(const float* __restrict__ in,
                                   bf16_t* __restrict__ out, int n) {
    int i = blockIdx.x * blockDim.x + threadIdx.x;
    if (i < n) out[i] = (bf16_t)in[i];
}

// convert + gate-interleave permutation of a (2048 x K) weight matrix:
// new row n' = w*64 + gate*16 + q  <-  old row gate*512 + w*16 + q
__global__ void pack_weight_kernel(const float* __restrict__ W,
                                   bf16_t* __restrict__ Wp, int K) {
    int i = blockIdx.x * blockDim.x + threadIdx.x;
    if (i >= G_ * K) return;
    int np = i / K, k = i - np * K;
    int w = np >> 6, gate = (np >> 4) & 3, q = np & 15;
    int old = gate * 512 + w * 16 + q;
    Wp[i] = (bf16_t)W[(size_t)old * K + k];
}

__global__ void pack_bias_kernel(const float* a0, const float* b0,
                                 const float* a1, const float* b1,
                                 float* o0, float* o1) {
    int np = blockIdx.x * blockDim.x + threadIdx.x;
    if (np >= G_) return;
    int w = np >> 6, gate = (np >> 4) & 3, q = np & 15;
    int old = gate * 512 + w * 16 + q;
    o0[np] = a0[old] + b0[old];
    o1[np] = a1[old] + b1[old];
}

__global__ void fc_kernel(const bf16_t* __restrict__ h,   // (64,512) final h1
                          const float* __restrict__ W,    // (24,512)
                          const float* __restrict__ bias, // (24)
                          float* __restrict__ out) {      // (64,24)
    int idx = blockIdx.x * blockDim.x + threadIdx.x;
    if (idx >= B_ * OUT_) return;
    int b = idx / OUT_, o = idx % OUT_;
    const bf16_t* hr = h + (size_t)b * H_;
    const float*  wr = W + (size_t)o * H_;
    float acc = bias[o];
#pragma unroll 8
    for (int k = 0; k < H_; ++k) acc += (float)hr[k] * wr[k];
    out[idx] = acc;
}

// ---------------- launch ---------------------------------------------------
extern "C" void kernel_launch(void* const* d_in, const int* in_sizes, int n_in,
                              void* d_out, int out_size, void* d_ws, size_t ws_size,
                              hipStream_t stream) {
    (void)in_sizes; (void)n_in; (void)out_size; (void)ws_size;
    const float* x    = (const float*)d_in[0];
    const float* Wih0 = (const float*)d_in[1];
    const float* Whh0 = (const float*)d_in[2];
    const float* bih0 = (const float*)d_in[3];
    const float* bhh0 = (const float*)d_in[4];
    const float* Wih1 = (const float*)d_in[5];
    const float* Whh1 = (const float*)d_in[6];
    const float* bih1 = (const float*)d_in[7];
    const float* bhh1 = (const float*)d_in[8];
    const float* fcW  = (const float*)d_in[9];
    const float* fcb  = (const float*)d_in[10];

    // workspace carve-out (~15.5 MB total)
    size_t off = 0;
    char* base = (char*)d_ws;
    auto take = [&](size_t bytes) -> void* {
        void* p = base + off;
        off += (bytes + 255) & ~(size_t)255;
        return p;
    };
    bf16_t* xb    = (bf16_t*)take((size_t)B_ * T_ * I_ * 2);  // 8 MB
    bf16_t* Wih0p = (bf16_t*)take((size_t)G_ * I_ * 2);
    bf16_t* Whh0p = (bf16_t*)take((size_t)G_ * H_ * 2);
    bf16_t* Wih1p = (bf16_t*)take((size_t)G_ * H_ * 2);
    bf16_t* Whh1p = (bf16_t*)take((size_t)G_ * H_ * 2);
    float*  bias0p = (float*)take(G_ * 4);
    float*  bias1p = (float*)take(G_ * 4);
    size_t zoff = off;                                  // zero-init region
    bf16_t* h0b = (bf16_t*)take((size_t)B_ * H_ * 2);
    float*  c0b = (float*)take((size_t)B_ * H_ * 4);
    bf16_t* h1b = (bf16_t*)take((size_t)B_ * H_ * 2);
    float*  c1b = (float*)take((size_t)B_ * H_ * 4);
    unsigned* bar = (unsigned*)take(256);
    hipMemsetAsync(base + zoff, 0, off - zoff, stream); // h/c states + barrier

    // fp32 -> bf16 conversion (x) and convert+permute (weights, biases)
    int nx = B_ * T_ * I_;
    f32_to_bf16_kernel<<<(nx + 255) / 256, 256, 0, stream>>>(x, xb, nx);
    pack_weight_kernel<<<(G_ * I_ + 255) / 256, 256, 0, stream>>>(Wih0, Wih0p, I_);
    pack_weight_kernel<<<(G_ * H_ + 255) / 256, 256, 0, stream>>>(Whh0, Whh0p, H_);
    pack_weight_kernel<<<(G_ * H_ + 255) / 256, 256, 0, stream>>>(Wih1, Wih1p, H_);
    pack_weight_kernel<<<(G_ * H_ + 255) / 256, 256, 0, stream>>>(Whh1, Whh1p, H_);
    pack_bias_kernel<<<8, 256, 0, stream>>>(bih0, bhh0, bih1, bhh1, bias0p, bias1p);

    // fused persistent 2-layer LSTM over all 512 timesteps
    lstm_persistent<<<NWG, BDIM, 0, stream>>>(xb, Wih0p, Whh0p, Wih1p, Whh1p,
                                              bias0p, bias1p,
                                              h0b, c0b, h1b, c1b, bar);

    // final projection: out[b][o] from last hidden state of layer 1
    fc_kernel<<<(B_ * OUT_ + 255) / 256, 256, 0, stream>>>(h1b, fcW, fcb,
                                                           (float*)d_out);
}
